// LinearRTUs_8993661518099
// MI455X (gfx1250) — matrix operand order repaired and checked by reading, not compiled, not run on hardware
//
#include <hip/hip_runtime.h>
#include <hip/hip_bf16.h>
#include <math.h>

// MI455X / gfx1250, wave32.
// Pass 1: split fp32 -> bf16(hi)+bf16(lo) for x, w1, w2 into ws (fragment order).
// Pass 2: LDS-blocked WMMA GEMM: block tile 128x64, 8 waves, each wave 32x32
//         of BOTH GEMM outputs; double-buffered 32KB/k-step LDS staging;
//         24 v_wmma per wave per k-step; fused rotation/norm/ReLU epilogue.
// Fallback (small ws): fused single-pass kernel with in-register splitting.

typedef __attribute__((ext_vector_type(16))) __bf16 v16bf;
typedef __attribute__((ext_vector_type(8)))  float  v8f;

#define B_DIM 16384
#define D_DIM 1024
#define H_DIM 1024

// ws element offsets (in __bf16 units)
#define XHI_OFF  ((size_t)0)
#define XLO_OFF  ((size_t)16 * 1024 * 1024)   // 1024 mt * 32 kb * 512
#define W1HI_OFF ((size_t)32 * 1024 * 1024)
#define W1LO_OFF ((size_t)33 * 1024 * 1024)
#define W2HI_OFF ((size_t)34 * 1024 * 1024)
#define W2LO_OFF ((size_t)35 * 1024 * 1024)
#define WS_NEED_BYTES ((size_t)36 * 1024 * 1024 * 2)   // 72 MB

#define BSPLIT(vh, vl, i, fval)                         \
  {                                                     \
    float _f = (fval);                                  \
    __bf16 _h = (__bf16)_f;                             \
    (vh)[i] = _h;                                       \
    (vl)[i] = (__bf16)(_f - (float)_h);                 \
  }

#define WMMA_BF16(A, Bm, C)                                              \
  __builtin_amdgcn_wmma_f32_16x16x32_bf16(false, (A), false, (Bm),       \
                                          (short)0, (C), false, false)

// ---------------------------------------------------------------------------
// Pass 1a: x (B x D fp32) -> A-fragment split bf16 (lane-major, 32B/lane).
// ---------------------------------------------------------------------------
__global__ __launch_bounds__(256) void cvt_x_kernel(const float* __restrict__ x,
                                                    __bf16* __restrict__ ws)
{
  const size_t t  = (size_t)blockIdx.x * 256 + threadIdx.x;  // [0, 1024*32*32)
  const int lane  = (int)(t & 31);
  const int kb    = (int)((t >> 5) & 31);
  const int mt    = (int)(t >> 10);

  const int row = mt * 16 + (lane & 15);
  const int k0  = kb * 32 + ((lane >> 4) << 3);
  const float* p = x + (size_t)row * D_DIM + k0;

  float4 a0 = *(const float4*)(p);
  float4 a1 = *(const float4*)(p + 4);
  float4 a2 = *(const float4*)(p + 16);
  float4 a3 = *(const float4*)(p + 20);

  v16bf hi, lo;
  BSPLIT(hi, lo, 0,  a0.x) BSPLIT(hi, lo, 1,  a0.y)
  BSPLIT(hi, lo, 2,  a0.z) BSPLIT(hi, lo, 3,  a0.w)
  BSPLIT(hi, lo, 4,  a1.x) BSPLIT(hi, lo, 5,  a1.y)
  BSPLIT(hi, lo, 6,  a1.z) BSPLIT(hi, lo, 7,  a1.w)
  BSPLIT(hi, lo, 8,  a2.x) BSPLIT(hi, lo, 9,  a2.y)
  BSPLIT(hi, lo, 10, a2.z) BSPLIT(hi, lo, 11, a2.w)
  BSPLIT(hi, lo, 12, a3.x) BSPLIT(hi, lo, 13, a3.y)
  BSPLIT(hi, lo, 14, a3.z) BSPLIT(hi, lo, 15, a3.w)

  *(v16bf*)(ws + XHI_OFF + t * 16) = hi;
  *(v16bf*)(ws + XLO_OFF + t * 16) = lo;
}

// ---------------------------------------------------------------------------
// Pass 1b: w1/w2 (D x H fp32) -> B-fragment split bf16.
// ---------------------------------------------------------------------------
__global__ __launch_bounds__(256) void cvt_w_kernel(const float* __restrict__ w1,
                                                    const float* __restrict__ w2,
                                                    __bf16* __restrict__ ws)
{
  const size_t t  = (size_t)blockIdx.x * 256 + threadIdx.x;  // [0, 2*64*32*32)
  const size_t u  = t & 65535;
  const int mat   = (int)(t >> 16);          // 0 = w1, 1 = w2
  const int lane  = (int)(u & 31);
  const int kb    = (int)((u >> 5) & 31);
  const int nt    = (int)(u >> 10);

  const int col   = nt * 16 + (lane & 15);
  const int kbase = kb * 32 + ((lane >> 4) << 3);
  const float* w  = mat ? w2 : w1;
  const float* p  = w + (size_t)kbase * H_DIM + col;

  v16bf hi, lo;
#pragma unroll
  for (int i = 0; i < 16; ++i) {
    float v = p[(size_t)(i < 8 ? i : i + 8) * H_DIM];
    BSPLIT(hi, lo, i, v)
  }

  const size_t hoff = mat ? W2HI_OFF : W1HI_OFF;
  const size_t loff = mat ? W2LO_OFF : W1LO_OFF;
  *(v16bf*)(ws + hoff + u * 16) = hi;
  *(v16bf*)(ws + loff + u * 16) = lo;
}

// ---------------------------------------------------------------------------
// Pass 2: LDS-blocked GEMM. Block tile M=128 x N=64 (both GEMMs), 8 waves
// in 4(m) x 2(n); wave tile 32x32. Double-buffered LDS staging of 32KB per
// K=32 step. 24 v_wmma_f32_16x16x32_bf16 per wave per k-step.
// ---------------------------------------------------------------------------
__global__ __launch_bounds__(256) void gemm_blocked_kernel(
    const __bf16* __restrict__ ws,
    const float* __restrict__ h_c1,
    const float* __restrict__ h_c2,
    const float* __restrict__ r_param,
    const float* __restrict__ th_param,
    float* __restrict__ out)
{
  // sA: [buf][(mt_i*2 + hl)*512 + lane*16 + e]          (8 m-tiles x hi/lo)
  // sB: [buf][(((nt_i*2 + mat)*2) + hl)*512 + ...]      (4 n-tiles x 2 mats x hi/lo)
  __shared__ __bf16 sA[2][16 * 512];   // 16 KB per buffer
  __shared__ __bf16 sB[2][16 * 512];   // 16 KB per buffer

  const int tid  = threadIdx.x;
  const int lane = tid & 31;
  const int wave = tid >> 5;
  const int wy   = wave >> 1;          // 0..3 (m)
  const int wx   = wave & 1;           // 0..1 (n)

  const int bn  = blockIdx.x & 15;     // H/64 = 16 blocks along N
  const int bm  = blockIdx.x >> 4;     // B/128 = 128 blocks along M
  const int mt0 = bm * 8;              // base 16-row m-tile
  const int nt0 = bn * 4;              // base 16-col n-tile

  // ---- staging decomposition: 16 chunks of 1KB each for A and for B;
  //      16 threads per chunk, 2 x uint4 (32B) per thread.
  const int chunk = tid >> 4;          // 0..15
  const int sub   = tid & 15;          // 0..15
  const int a_mt  = chunk >> 1;
  const int a_hl  = chunk & 1;
  const int b_nt  = chunk >> 2;
  const int b_mat = (chunk >> 1) & 1;
  const int b_hl  = chunk & 1;

  const __bf16* aG = ws + (size_t)a_hl * XLO_OFF +
                     ((size_t)(mt0 + a_mt) * 32) * 512;
  const __bf16* bG = ws + W1HI_OFF + (size_t)b_mat * (2u * 1024 * 1024) +
                     (size_t)b_hl * (1024 * 1024) +
                     ((size_t)(nt0 + b_nt) * 32) * 512;

  auto stage = [&](int kb, int buf) {
    const uint4* ga = (const uint4*)(aG + (size_t)kb * 512);
    const uint4* gb = (const uint4*)(bG + (size_t)kb * 512);
    uint4* la = (uint4*)&sA[buf][chunk * 512];
    uint4* lb = (uint4*)&sB[buf][chunk * 512];
    la[sub * 2]     = ga[sub * 2];
    la[sub * 2 + 1] = ga[sub * 2 + 1];
    lb[sub * 2]     = gb[sub * 2];
    lb[sub * 2 + 1] = gb[sub * 2 + 1];
  };

  v8f acc[2][2][2];                    // [gemm][mi][ni]
#pragma unroll
  for (int g = 0; g < 2; ++g)
#pragma unroll
    for (int mi = 0; mi < 2; ++mi)
#pragma unroll
      for (int ni = 0; ni < 2; ++ni)
        acc[g][mi][ni] = (v8f){0.f, 0.f, 0.f, 0.f, 0.f, 0.f, 0.f, 0.f};

  const int mbase = wy * 2;
  const int nbase = wx * 2;
  const int lof   = lane * 16;

  stage(0, 0);
  __syncthreads();

  for (int kb = 0; kb < 32; ++kb) {
    const int buf = kb & 1;
    if (kb + 1 < 32) stage(kb + 1, buf ^ 1);

    v16bf a0h = *(const v16bf*)&sA[buf][((mbase + 0) * 2 + 0) * 512 + lof];
    v16bf a0l = *(const v16bf*)&sA[buf][((mbase + 0) * 2 + 1) * 512 + lof];
    v16bf a1h = *(const v16bf*)&sA[buf][((mbase + 1) * 2 + 0) * 512 + lof];
    v16bf a1l = *(const v16bf*)&sA[buf][((mbase + 1) * 2 + 1) * 512 + lof];

#pragma unroll
    for (int ni = 0; ni < 2; ++ni) {
      const int nc = nbase + ni;
      v16bf b1h = *(const v16bf*)&sB[buf][((nc * 2 + 0) * 2 + 0) * 512 + lof];
      v16bf b1l = *(const v16bf*)&sB[buf][((nc * 2 + 0) * 2 + 1) * 512 + lof];
      v16bf b2h = *(const v16bf*)&sB[buf][((nc * 2 + 1) * 2 + 0) * 512 + lof];
      v16bf b2l = *(const v16bf*)&sB[buf][((nc * 2 + 1) * 2 + 1) * 512 + lof];

#pragma unroll
      for (int mi = 0; mi < 2; ++mi) {
        v16bf ah = mi ? a1h : a0h;
        v16bf al = mi ? a1l : a0l;
        acc[0][mi][ni] = WMMA_BF16(ah, b1h, acc[0][mi][ni]);
        acc[0][mi][ni] = WMMA_BF16(ah, b1l, acc[0][mi][ni]);
        acc[0][mi][ni] = WMMA_BF16(al, b1h, acc[0][mi][ni]);
        acc[1][mi][ni] = WMMA_BF16(ah, b2h, acc[1][mi][ni]);
        acc[1][mi][ni] = WMMA_BF16(ah, b2l, acc[1][mi][ni]);
        acc[1][mi][ni] = WMMA_BF16(al, b2h, acc[1][mi][ni]);
      }
    }
    __syncthreads();
  }

  // ---- fused epilogue. C layout: elem r -> M = r + 8*(lane>>4), N = lane&15.
  const int madd = (lane >> 4) << 3;
  float* out_c1 = out;
  float* out_c2 = out + (size_t)B_DIM * H_DIM;
  float* out_ht = out + (size_t)2 * B_DIM * H_DIM;

#pragma unroll
  for (int ni = 0; ni < 2; ++ni) {
    const int col = (nt0 + nbase + ni) * 16 + (lane & 15);
    float rp  = r_param[col];
    float th  = th_param[col];
    float g   = rp * cosf(th);
    float ph  = rp * sinf(th);
    float nrm = sqrtf(fmaxf(1.0f - rp * rp, 0.0f));

#pragma unroll
    for (int mi = 0; mi < 2; ++mi) {
      const int rowbase = (mt0 + mbase + mi) * 16 + madd;
#pragma unroll
      for (int r = 0; r < 8; ++r) {
        const int row = rowbase + r;
        const size_t idx = (size_t)row * H_DIM + col;
        float c1 = h_c1[idx];
        float c2 = h_c2[idx];
        float o1 = g * c1 - ph * c2 + nrm * acc[0][mi][ni][r];
        float o2 = g * c2 + ph * c1 + nrm * acc[1][mi][ni][r];
        out_c1[idx] = o1;
        out_c2[idx] = o2;
        const size_t hrow = (size_t)row * (2 * H_DIM);
        out_ht[hrow + col]         = fmaxf(o1, 0.0f);
        out_ht[hrow + H_DIM + col] = fmaxf(o2, 0.0f);
      }
    }
  }
}

// ---------------------------------------------------------------------------
// Fallback: fused single-pass kernel (in-register splitting), if ws too small.
// ---------------------------------------------------------------------------
__global__ __launch_bounds__(256) void rtus_fused_kernel(
    const float* __restrict__ h_c1,
    const float* __restrict__ h_c2,
    const float* __restrict__ x,
    const float* __restrict__ r_param,
    const float* __restrict__ th_param,
    const float* __restrict__ w1,
    const float* __restrict__ w2,
    float* __restrict__ out)
{
  const int lane = threadIdx.x & 31;
  const int wave = threadIdx.x >> 5;
  const int tile = blockIdx.x * 8 + wave;
  const int NT   = H_DIM / 16;
  const int mt   = tile / NT;
  const int nt   = tile - mt * NT;
  const int m0   = mt * 16;
  const int n0   = nt * 16;

  const int arow = lane & 15;
  const int akb  = (lane >> 4) << 3;
  const float* aptr = x + (size_t)(m0 + arow) * D_DIM + akb;

  const int bcol = lane & 15;
  const int bkb  = (lane >> 4) << 3;
  const float* b1ptr = w1 + (size_t)bkb * H_DIM + (n0 + bcol);
  const float* b2ptr = w2 + (size_t)bkb * H_DIM + (n0 + bcol);

  v8f acc1 = {0.f, 0.f, 0.f, 0.f, 0.f, 0.f, 0.f, 0.f};
  v8f acc2 = {0.f, 0.f, 0.f, 0.f, 0.f, 0.f, 0.f, 0.f};

  for (int k0 = 0; k0 < D_DIM; k0 += 32) {
    float4 a0 = *(const float4*)(aptr + k0);
    float4 a1 = *(const float4*)(aptr + k0 + 4);
    float4 a2 = *(const float4*)(aptr + k0 + 16);
    float4 a3 = *(const float4*)(aptr + k0 + 20);
    v16bf ahi, alo;
    BSPLIT(ahi, alo, 0,  a0.x) BSPLIT(ahi, alo, 1,  a0.y)
    BSPLIT(ahi, alo, 2,  a0.z) BSPLIT(ahi, alo, 3,  a0.w)
    BSPLIT(ahi, alo, 4,  a1.x) BSPLIT(ahi, alo, 5,  a1.y)
    BSPLIT(ahi, alo, 6,  a1.z) BSPLIT(ahi, alo, 7,  a1.w)
    BSPLIT(ahi, alo, 8,  a2.x) BSPLIT(ahi, alo, 9,  a2.y)
    BSPLIT(ahi, alo, 10, a2.z) BSPLIT(ahi, alo, 11, a2.w)
    BSPLIT(ahi, alo, 12, a3.x) BSPLIT(ahi, alo, 13, a3.y)
    BSPLIT(ahi, alo, 14, a3.z) BSPLIT(ahi, alo, 15, a3.w)

    v16bf b1h, b1l, b2h, b2l;
#pragma unroll
    for (int i = 0; i < 16; ++i) {
      float b1v = b1ptr[(size_t)(k0 + (i < 8 ? i : i + 8)) * H_DIM];
      float b2v = b2ptr[(size_t)(k0 + (i < 8 ? i : i + 8)) * H_DIM];
      BSPLIT(b1h, b1l, i, b1v)
      BSPLIT(b2h, b2l, i, b2v)
    }

    acc1 = WMMA_BF16(ahi, b1h, acc1);
    acc1 = WMMA_BF16(ahi, b1l, acc1);
    acc1 = WMMA_BF16(alo, b1h, acc1);
    acc2 = WMMA_BF16(ahi, b2h, acc2);
    acc2 = WMMA_BF16(ahi, b2l, acc2);
    acc2 = WMMA_BF16(alo, b2h, acc2);
  }

  const int col  = n0 + (lane & 15);
  const int madd = (lane >> 4) << 3;

  float rp  = r_param[col];
  float th  = th_param[col];
  float g   = rp * cosf(th);
  float ph  = rp * sinf(th);
  float nrm = sqrtf(fmaxf(1.0f - rp * rp, 0.0f));

  float* out_c1 = out;
  float* out_c2 = out + (size_t)B_DIM * H_DIM;
  float* out_ht = out + (size_t)2 * B_DIM * H_DIM;

#pragma unroll
  for (int r = 0; r < 8; ++r) {
    int row = m0 + r + madd;
    size_t idx = (size_t)row * H_DIM + col;
    float c1 = h_c1[idx];
    float c2 = h_c2[idx];
    float o1 = g * c1 - ph * c2 + nrm * acc1[r];
    float o2 = g * c2 + ph * c1 + nrm * acc2[r];
    out_c1[idx] = o1;
    out_c2[idx] = o2;
    size_t hrow = (size_t)row * (2 * H_DIM);
    out_ht[hrow + col]         = fmaxf(o1, 0.0f);
    out_ht[hrow + H_DIM + col] = fmaxf(o2, 0.0f);
  }
}

extern "C" void kernel_launch(void* const* d_in, const int* in_sizes, int n_in,
                              void* d_out, int out_size, void* d_ws, size_t ws_size,
                              hipStream_t stream) {
  (void)in_sizes; (void)n_in; (void)out_size;
  const float* h_c1 = (const float*)d_in[0];
  const float* h_c2 = (const float*)d_in[1];
  const float* x_t  = (const float*)d_in[2];
  const float* r_p  = (const float*)d_in[3];
  const float* th_p = (const float*)d_in[4];
  const float* w1   = (const float*)d_in[5];
  const float* w2   = (const float*)d_in[6];
  float* out = (float*)d_out;

  if (ws_size >= WS_NEED_BYTES && d_ws != nullptr) {
    __bf16* ws = (__bf16*)d_ws;
    // Pass 1: one-time split conversion into fragment layout.
    cvt_x_kernel<<<(1024 * 32 * 32) / 256, 256, 0, stream>>>(x_t, ws);
    cvt_w_kernel<<<(2 * 64 * 32 * 32) / 256, 256, 0, stream>>>(w1, w2, ws);
    // Pass 2: LDS-blocked WMMA GEMM + fused epilogue.
    // Grid: (B/128) x (H/64) = 128 x 16 = 2048 blocks of 256 threads.
    gemm_blocked_kernel<<<2048, 256, 0, stream>>>(ws, h_c1, h_c2, r_p, th_p, out);
  } else {
    const int tiles  = (B_DIM / 16) * (H_DIM / 16);
    const int blocks = tiles / 8;
    rtus_fused_kernel<<<blocks, 256, 0, stream>>>(h_c1, h_c2, x_t, r_p, th_p,
                                                  w1, w2, out);
  }
}
